// simpleGraphConvolutionalignment_53163105190093
// MI455X (gfx1250) — matrix operand order repaired and checked by reading, not compiled, not run on hardware
//
#include <hip/hip_runtime.h>
#include <hip/hip_bf16.h>
#include <math.h>

// ---------------------------------------------------------------------------
// Problem constants (from reference): B=8, N=1024, H=200, K=3 layers.
// H padded to HP=224 (multiple of 32 for bf16 WMMA K-depth). The zero-pad
// invariant for columns [200,224) is maintained by every epilogue.
// ---------------------------------------------------------------------------
constexpr int Bn = 8;
constexpr int Nn = 1024;
constexpr int Hn = 200;
constexpr int HP = 224;
constexpr int Kl = 3;

constexpr int BM = 128;   // block tile M
constexpr int BNt = 128;  // block tile N
constexpr int BK = 32;    // block tile K (== bf16 WMMA K)
constexpr int SAP = 36;   // f32 A-staging row stride (16B-aligned, bank-safe)

typedef __attribute__((ext_vector_type(16))) __bf16 v16bf;
typedef __attribute__((ext_vector_type(8)))  float  v8f;
typedef int v4i_t __attribute__((__vector_size__(4 * sizeof(int))));

union FragU { v16bf v; unsigned int u[8]; __bf16 e[16]; };

// CDNA5 async global->LDS path (ASYNCcnt), guarded so the kernel compiles
// on toolchains without the builtins (falls back to b128 sync staging).
#if defined(__gfx1250__) && __has_builtin(__builtin_amdgcn_global_load_async_to_lds_b128) && __has_builtin(__builtin_amdgcn_s_wait_asynccnt)
#define USE_ASYNC_A 1
#else
#define USE_ASYNC_A 0
#endif

__device__ __forceinline__ unsigned short f2bf(float f) {
  // native f32 -> bf16 convert (RNE); single v_cvt on gfx1250
  return __builtin_bit_cast(unsigned short, (__bf16)f);
}

enum EpiKind {
  EPI_PLAIN = 0,   // C = acc                        (pad cols forced 0)
  EPI_BIAS,        // C = acc + bias[col]
  EPI_SCORES,      // C = acc*scale + maskAdd[b,col]
  EPI_PAIR,        // C = acc + (1 - tm[b,row]*tm[b,col]) * -1e4
  EPI_BIAS_RES,    // C = acc + bias[col] + res
  EPI_BIAS_GELU,   // C = gelu(acc + bias[col])      (exact erf gelu)
  EPI_GCN,         // C = relu(acc / denom[b,row]) + res
  EPI_ALIGN        // C = acc * tm[b,row] + res
};

// ---------------------------------------------------------------------------
// Generic batched WMMA GEMM:  C[b] = epilogue( A[b] @ B[b or shared] )
// A: f32 [1024, Kpad] row-major (always full tiles, 16B-aligned rows).
// B: f32; TB -> element (k,n) read from B[n*ldb+k] (A @ B^T).
// BVEC -> B is a padded activation tensor: unguarded float4 staging
//         (over-read columns feed only discarded (col>=Cw) outputs).
// Double-buffered LDS, one barrier per K-step. A tile staged raw f32
// (async b128 if available), converted to bf16 while building fragments;
// B staged f32->bf16 via VGPRs with stores placed after the fragment loads
// so DS in-order issue does not stall the WMMA stream on global latency.
// 256 threads = 8 waves; wave tile 32x64 = 2x4 v_wmma_f32_16x16x32_bf16.
// ---------------------------------------------------------------------------
template<int EPI, bool TB, bool BVEC>
__global__ __launch_bounds__(256) void gemm_kernel(
    const float* __restrict__ A, long long strideA, int lda,
    const float* __restrict__ Bm, long long strideB, int ldb,
    float* __restrict__ C, long long strideC, int ldc,
    int Cw, int Ncols, int Kdim, int Ksteps,
    const float* __restrict__ bias,
    const float* __restrict__ res, long long strideRes,
    const float* __restrict__ aux,
    float scale)
{
  __shared__ float          sAf[2][BM][SAP];       // raw f32 A tiles (double)
  __shared__ unsigned short sBt[2][BK][BNt + 2];   // bf16 B tiles    (double)

  const int tid  = threadIdx.x;
  const int bz   = blockIdx.z;
  const int row0 = blockIdx.y * BM;
  const int col0 = blockIdx.x * BNt;

  const float* Ab = A + (size_t)bz * strideA;
  const float* Bb = Bm + (size_t)bz * strideB;
  float*       Cb = C + (size_t)bz * strideC;

  const int lane = tid & 31;
  const int h    = lane >> 4;   // lane half
  const int r    = lane & 15;
  const int wid  = tid >> 5;
  const int wm   = wid & 3;     // 4 wave rows  -> 32 rows each
  const int wn   = wid >> 2;    // 2 wave cols  -> 64 cols each

  // A staging chunk coords: 16B chunks; thread covers rows am0+{0,32,64,96}
  const int am0 = tid >> 3;
  const int ac4 = (tid & 7) * 4;

  // A-fragment K indices per ISA 16-bit A 16x32 table
  int kkA[8];
#pragma unroll
  for (int j = 0; j < 8; ++j)
    kkA[j] = (j < 4) ? (8 * h + 2 * j) : (16 + 8 * h + 2 * (j - 4));
  // B-fragment: lane -> N column r, VGPR j -> K pair (kkA[j], kkA[j]+1); kRow is unused
  const int kRow = r + 16 * h;

  v8f acc[2][4] = {};

  // ---- prologue: stage tile 0 into buffer 0 ----
  {
    const int k0 = 0;
#if USE_ASYNC_A
#pragma unroll
    for (int i = 0; i < 4; ++i) {
      const int m = am0 + i * 32;
      const float* g = Ab + (size_t)(row0 + m) * lda + k0 + ac4;
      __builtin_amdgcn_global_load_async_to_lds_b128(
          (__attribute__((address_space(1))) v4i_t*)g,
          (__attribute__((address_space(3))) v4i_t*)&sAf[0][m][ac4], 0, 0);
    }
#else
#pragma unroll
    for (int i = 0; i < 4; ++i) {
      const int m = am0 + i * 32;
      *(float4*)&sAf[0][m][ac4] =
          *(const float4*)(Ab + (size_t)(row0 + m) * lda + k0 + ac4);
    }
#endif
    if (BVEC) {
#pragma unroll
      for (int i = 0; i < 4; ++i) {
        int idx = tid + i * 256;
        float4 t;
        if (TB) {
          int n = idx >> 3, c4 = (idx & 7) * 4;
          t = *(const float4*)(Bb + (size_t)(col0 + n) * ldb + k0 + c4);
#pragma unroll
          for (int q = 0; q < 4; ++q)
            sBt[0][c4 + q][n] = f2bf(((const float*)&t)[q]);
        } else {
          int kk = idx >> 5, c4 = (idx & 31) * 4;
          t = *(const float4*)(Bb + (size_t)(k0 + kk) * ldb + col0 + c4);
          *(unsigned int*)&sBt[0][kk][c4] =
              (unsigned)f2bf(t.x) | ((unsigned)f2bf(t.y) << 16);
          *(unsigned int*)&sBt[0][kk][c4 + 2] =
              (unsigned)f2bf(t.z) | ((unsigned)f2bf(t.w) << 16);
        }
      }
    } else {
#pragma unroll
      for (int i = 0; i < 16; ++i) {
        int idx = tid + i * 256;
        int kk = TB ? (idx & 31) : (idx >> 7);
        int n  = TB ? (idx >> 5) : (idx & 127);
        int kg = k0 + kk, nc = col0 + n;
        float v = 0.0f;
        if (kg < Kdim && nc < Ncols)
          v = TB ? Bb[(size_t)nc * ldb + kg] : Bb[(size_t)kg * ldb + nc];
        sBt[0][kk][n] = f2bf(v);
      }
    }
  }

  int sel = 0;
  for (int kt = 0; kt < Ksteps; ++kt) {
#if USE_ASYNC_A
    __builtin_amdgcn_s_wait_asynccnt(0);   // tile `sel` async copies done
#endif
    __syncthreads();                       // publish tile `sel` to all waves

    const bool next = (kt + 1 < Ksteps);
    const int  k0n  = (kt + 1) * BK;
    const int  ns   = sel ^ 1;

    // ---- issue next tile's fetches before computing current tile ----
#if USE_ASYNC_A
    if (next) {
#pragma unroll
      for (int i = 0; i < 4; ++i) {
        const int m = am0 + i * 32;
        const float* g = Ab + (size_t)(row0 + m) * lda + k0n + ac4;
        __builtin_amdgcn_global_load_async_to_lds_b128(
            (__attribute__((address_space(1))) v4i_t*)g,
            (__attribute__((address_space(3))) v4i_t*)&sAf[ns][m][ac4], 0, 0);
      }
    }
#else
    float4 areg[4];
    if (next) {
      __builtin_prefetch(Ab + (size_t)(row0 + am0) * lda + k0n + BK + ac4, 0, 1);
#pragma unroll
      for (int i = 0; i < 4; ++i)
        areg[i] = *(const float4*)(Ab + (size_t)(row0 + am0 + i * 32) * lda + k0n + ac4);
    }
#endif
    float breg[16];
    if (next) {
      if (BVEC) {
#pragma unroll
        for (int i = 0; i < 4; ++i) {
          int idx = tid + i * 256;
          const float* p;
          if (TB) { int n = idx >> 3, c4 = (idx & 7) * 4;
                    p = Bb + (size_t)(col0 + n) * ldb + k0n + c4; }
          else    { int kk = idx >> 5, c4 = (idx & 31) * 4;
                    p = Bb + (size_t)(k0n + kk) * ldb + col0 + c4; }
          float4 t = *(const float4*)p;
          breg[i * 4 + 0] = t.x; breg[i * 4 + 1] = t.y;
          breg[i * 4 + 2] = t.z; breg[i * 4 + 3] = t.w;
        }
      } else {
#pragma unroll
        for (int i = 0; i < 16; ++i) {
          int idx = tid + i * 256;
          int kk = TB ? (idx & 31) : (idx >> 7);
          int n  = TB ? (idx >> 5) : (idx & 127);
          int kg = k0n + kk, nc = col0 + n;
          float v = 0.0f;
          if (kg < Kdim && nc < Ncols)
            v = TB ? Bb[(size_t)nc * ldb + kg] : Bb[(size_t)kg * ldb + nc];
          breg[i] = v;
        }
      }
    }

    // ---- compute current tile: fragments + 8 WMMA per wave ----
    FragU a[2], b[4];
#pragma unroll
    for (int tmi = 0; tmi < 2; ++tmi) {
      const int mrow = wm * 32 + tmi * 16 + r;
#pragma unroll
      for (int j = 0; j < 8; ++j) {
        float2 p = *(const float2*)&sAf[sel][mrow][kkA[j]];
        a[tmi].e[2 * j]     = (__bf16)p.x;
        a[tmi].e[2 * j + 1] = (__bf16)p.y;
      }
    }
#pragma unroll
    for (int tni = 0; tni < 4; ++tni)
#pragma unroll
      for (int j = 0; j < 8; ++j) {
        b[tni].e[2 * j]     = __builtin_bit_cast(__bf16, sBt[sel][kkA[j]][wn * 64 + tni * 16 + r]);
        b[tni].e[2 * j + 1] = __builtin_bit_cast(__bf16, sBt[sel][kkA[j] + 1][wn * 64 + tni * 16 + r]);
      }
#pragma unroll
    for (int tmi = 0; tmi < 2; ++tmi)
#pragma unroll
      for (int tni = 0; tni < 4; ++tni)
        acc[tmi][tni] = __builtin_amdgcn_wmma_f32_16x16x32_bf16(
            false, a[tmi].v, false, b[tni].v, (short)0, acc[tmi][tni], false, false);

    // ---- store next tile's staged data (after frag ds-loads: DS in-order) ----
    if (next) {
#if !USE_ASYNC_A
#pragma unroll
      for (int i = 0; i < 4; ++i)
        *(float4*)&sAf[ns][am0 + i * 32][ac4] = areg[i];
#endif
      if (BVEC) {
#pragma unroll
        for (int i = 0; i < 4; ++i) {
          int idx = tid + i * 256;
          if (TB) {
            int n = idx >> 3, c4 = (idx & 7) * 4;
#pragma unroll
            for (int q = 0; q < 4; ++q) sBt[ns][c4 + q][n] = f2bf(breg[i * 4 + q]);
          } else {
            int kk = idx >> 5, c4 = (idx & 31) * 4;
            *(unsigned int*)&sBt[ns][kk][c4] =
                (unsigned)f2bf(breg[i * 4 + 0]) | ((unsigned)f2bf(breg[i * 4 + 1]) << 16);
            *(unsigned int*)&sBt[ns][kk][c4 + 2] =
                (unsigned)f2bf(breg[i * 4 + 2]) | ((unsigned)f2bf(breg[i * 4 + 3]) << 16);
          }
        }
      } else {
#pragma unroll
        for (int i = 0; i < 16; ++i) {
          int idx = tid + i * 256;
          int kk = TB ? (idx & 31) : (idx >> 7);
          int n  = TB ? (idx >> 5) : (idx & 127);
          sBt[ns][kk][n] = f2bf(breg[i]);
        }
      }
    }
    sel = ns;
  }

  // ---- epilogue: C/D layout -> VGPR i holds row (i + 8h), lane r = col ----
#pragma unroll
  for (int tmi = 0; tmi < 2; ++tmi) {
#pragma unroll
    for (int tni = 0; tni < 4; ++tni) {
      const int mB = wm * 32 + tmi * 16;
      const int nB = wn * 64 + tni * 16;
      const int col = col0 + nB + r;
      if (col >= Cw) continue;
#pragma unroll
      for (int i = 0; i < 8; ++i) {
        const int row = row0 + mB + i + 8 * h;
        float v = acc[tmi][tni][i];
        float outv;
        if (EPI == EPI_SCORES) {
          outv = v * scale + aux[(size_t)bz * Nn + col];
        } else if (EPI == EPI_PAIR) {
          float p = aux[(size_t)bz * Nn + row] * aux[(size_t)bz * Nn + col];
          outv = v + (1.0f - p) * -10000.0f;
        } else if (col >= Ncols) {
          outv = 0.0f;   // maintain zero-pad invariant for padded H columns
        } else if (EPI == EPI_PLAIN) {
          outv = v;
        } else if (EPI == EPI_BIAS) {
          outv = v + bias[col];
        } else if (EPI == EPI_BIAS_RES) {
          outv = v + bias[col] + res[(size_t)bz * strideRes + (size_t)row * ldc + col];
        } else if (EPI == EPI_BIAS_GELU) {
          float t2 = v + bias[col];
          outv = 0.5f * t2 * (1.0f + erff(t2 * 0.70710678118654752f));
        } else if (EPI == EPI_GCN) {
          outv = fmaxf(v / aux[(size_t)bz * Nn + row], 0.0f)
               + res[(size_t)bz * strideRes + (size_t)row * ldc + col];
        } else { // EPI_ALIGN
          outv = v * aux[(size_t)bz * Nn + row]
               + res[(size_t)bz * strideRes + (size_t)row * ldc + col];
        }
        Cb[(size_t)row * ldc + col] = outv;
      }
    }
  }
}

// ---------------------------------------------------------------------------
// Elementwise / reduction kernels
// ---------------------------------------------------------------------------
__global__ void prep_adj_kernel(const float* __restrict__ a1, const float* __restrict__ a2,
                                float* __restrict__ adj, float* __restrict__ denom) {
  __shared__ float red[256];
  size_t row = blockIdx.x;                    // b*N + n
  const float* p1 = a1 + row * Nn;
  const float* p2 = a2 + row * Nn;
  float* po = adj + row * Nn;
  int t = threadIdx.x;
  float s = 0.0f;
  for (int c = t; c < Nn; c += 256) {
    float v = p1[c] + p2[c];
    v = (v >= 1.0f) ? 1.0f : v;
    po[c] = v;
    s += v;
  }
  red[t] = s; __syncthreads();
  for (int st = 128; st > 0; st >>= 1) { if (t < st) red[t] += red[t + st]; __syncthreads(); }
  if (t == 0) denom[row] = red[0] + 1e-7f;
}

__global__ void prep_mask_kernel(const float* __restrict__ tm, float* __restrict__ maskAdd) {
  int idx = blockIdx.x * 256 + threadIdx.x;
  if (idx < Bn * Nn) maskAdd[idx] = (1.0f - tm[idx]) * -10000.0f;
}

__global__ void init_act_kernel(const float* __restrict__ text,
                                float* __restrict__ X, float* __restrict__ Y) {
  size_t idx = (size_t)blockIdx.x * 256 + threadIdx.x;
  if (idx >= (size_t)Bn * Nn * HP) return;
  size_t row = idx / HP;
  int c = (int)(idx % HP);
  float v = (c < Hn) ? text[row * Hn + c] : 0.0f;
  X[idx] = v;
  Y[idx] = v;
}

__global__ void softmax_row_kernel(float* __restrict__ buf) {
  __shared__ float red[256];
  size_t row = blockIdx.x;
  float* p = buf + row * (size_t)Nn;
  int t = threadIdx.x;
  float mx = -1e30f;
  for (int c = t; c < Nn; c += 256) mx = fmaxf(mx, p[c]);
  red[t] = mx; __syncthreads();
  for (int s = 128; s > 0; s >>= 1) { if (t < s) red[t] = fmaxf(red[t], red[t + s]); __syncthreads(); }
  mx = red[0]; __syncthreads();
  float sum = 0.0f;
  for (int c = t; c < Nn; c += 256) { float e = __expf(p[c] - mx); p[c] = e; sum += e; }
  red[t] = sum; __syncthreads();
  for (int s = 128; s > 0; s >>= 1) { if (t < s) red[t] += red[t + s]; __syncthreads(); }
  float inv = 1.0f / red[0];
  for (int c = t; c < Nn; c += 256) p[c] *= inv;
}

__global__ void transpose_kernel(const float* __restrict__ src, float* __restrict__ dst) {
  __shared__ float tile[32][33];
  const float* s = src + (size_t)blockIdx.z * Nn * Nn;
  float*       d = dst + (size_t)blockIdx.z * Nn * Nn;
  int x  = blockIdx.x * 32 + threadIdx.x;
  int y0 = blockIdx.y * 32;
  for (int i = 0; i < 32; i += 8)
    tile[threadIdx.y + i][threadIdx.x] = s[(size_t)(y0 + threadIdx.y + i) * Nn + x];
  __syncthreads();
  int x2 = blockIdx.y * 32 + threadIdx.x;
  int y2 = blockIdx.x * 32;
  for (int i = 0; i < 32; i += 8)
    d[(size_t)(y2 + threadIdx.y + i) * Nn + x2] = tile[threadIdx.x][threadIdx.y + i];
}

__global__ void layernorm_kernel(const float* __restrict__ src, float* __restrict__ dst,
                                 const float* __restrict__ g, const float* __restrict__ b) {
  __shared__ float red[256];
  size_t row = blockIdx.x;
  int t = threadIdx.x;
  float x = (t < Hn) ? src[row * HP + t] : 0.0f;
  red[t] = x; __syncthreads();
  for (int s = 128; s > 0; s >>= 1) { if (t < s) red[t] += red[t + s]; __syncthreads(); }
  float mu = red[0] * (1.0f / Hn); __syncthreads();
  float d = (t < Hn) ? (x - mu) : 0.0f;
  red[t] = d * d; __syncthreads();
  for (int s = 128; s > 0; s >>= 1) { if (t < s) red[t] += red[t + s]; __syncthreads(); }
  float var = red[0] * (1.0f / Hn);
  float inv = rsqrtf(var + 1e-20f);
  if (t < HP) dst[row * HP + t] = (t < Hn) ? (d * inv * g[t] + b[t]) : 0.0f;
}

__global__ void pack_out_kernel(const float* __restrict__ X, float* __restrict__ out) {
  size_t idx = (size_t)blockIdx.x * 256 + threadIdx.x;
  if (idx >= (size_t)Bn * Nn * Hn) return;
  size_t row = idx / Hn;
  int c = (int)(idx % Hn);
  out[idx] = X[row * HP + c];
}

// ---------------------------------------------------------------------------
// Host orchestration
// ---------------------------------------------------------------------------
template<int EPI, bool TB, bool BVEC>
static inline void launch_gemm(hipStream_t stream,
    const float* A, long long strideA, int lda,
    const float* Bm, long long strideB, int ldb,
    float* C, long long strideC, int ldc,
    int Cw, int Ncols, int Kdim, int Ksteps,
    const float* bias, const float* res, long long strideRes,
    const float* aux, float scale)
{
  dim3 grid((Cw + BNt - 1) / BNt, Nn / BM, Bn);
  gemm_kernel<EPI, TB, BVEC><<<grid, dim3(256), 0, stream>>>(
      A, strideA, lda, Bm, strideB, ldb, C, strideC, ldc,
      Cw, Ncols, Kdim, Ksteps, bias, res, strideRes, aux, scale);
}

extern "C" void kernel_launch(void* const* d_in, const int* in_sizes, int n_in,
                              void* d_out, int out_size, void* d_ws, size_t ws_size,
                              hipStream_t stream)
{
  (void)in_sizes; (void)n_in; (void)out_size;

  const float* text  = (const float*)d_in[0];
  const float* adj1  = (const float*)d_in[1];
  const float* adj2  = (const float*)d_in[2];
  const float* tmask = (const float*)d_in[5];
  const float* gcn_w = (const float*)d_in[6];
  const float* mut_w = (const float*)d_in[7];
  const float* qw = (const float*)d_in[8];   const float* qb = (const float*)d_in[9];
  const float* kw = (const float*)d_in[10];  const float* kb = (const float*)d_in[11];
  const float* vw = (const float*)d_in[12];  const float* vb = (const float*)d_in[13];
  const float* aow = (const float*)d_in[14]; const float* aob = (const float*)d_in[15];
  const float* g1 = (const float*)d_in[16];  const float* b1 = (const float*)d_in[17];
  const float* iw = (const float*)d_in[18];  const float* ib = (const float*)d_in[19];
  const float* ow = (const float*)d_in[20];  const float* ob = (const float*)d_in[21];
  const float* g2 = (const float*)d_in[22];  const float* b2 = (const float*)d_in[23];

  float* out_outs = (float*)d_out;
  float* out_adj  = out_outs + (size_t)Bn * Nn * Hn;   // output tuple: (outs, adj)

  const size_t NH  = (size_t)Bn * Nn * HP;   // padded activation tensor
  const size_t NNf = (size_t)Bn * Nn * Nn;   // [B,N,N] tensor
  const size_t needFloats = 7 * NH + 2 * NNf + 2 * (size_t)Bn * Nn;
  if (ws_size < needFloats * sizeof(float)) return;

  float* w  = (float*)d_ws;
  float* P0 = w;            // X  (outs)
  float* P1 = P0 + NH;      // Y  (output)
  float* P2 = P1 + NH;      // q / attn
  float* P3 = P2 + NH;      // k / Xb (bert outs)
  float* P4 = P3 + NH;      // v / teout
  float* P5 = P4 + NH;      // ctx / z2 / Yg
  float* P6 = P5 + NH;      // z / inter / t_mut
  float* S  = P6 + NH;      // [B,N,N] scores / logit / p_row
  float* ST = S + NNf;      // [B,N,N] logit^T / p_col^T
  float* maskAdd = ST + NNf;                  // [B,N]
  float* denom   = maskAdd + (size_t)Bn * Nn; // [B,N]

  // ---- prep: adj (into d_out, reused as GEMM input), denom, masks, init X/Y
  prep_adj_kernel<<<Bn * Nn, 256, 0, stream>>>(adj1, adj2, out_adj, denom);
  prep_mask_kernel<<<(Bn * Nn + 255) / 256, 256, 0, stream>>>(tmask, maskAdd);
  init_act_kernel<<<(unsigned)((NH + 255) / 256), 256, 0, stream>>>(text, P0, P1);

  const float scl = 0.07071067811865475f;  // 1/sqrt(200)
  const long long sAct = (long long)Nn * HP;
  const long long sNN  = (long long)Nn * Nn;
  const int KS_H = HP / BK;   // 7
  const int KS_N = Nn / BK;   // 32
  const size_t HH = (size_t)Hn * Hn;

  for (int l = 0; l < Kl; ++l) {
    const float* qw_l = qw + l * HH;  const float* qb_l = qb + (size_t)l * Hn;
    const float* kw_l = kw + l * HH;  const float* kb_l = kb + (size_t)l * Hn;
    const float* vw_l = vw + l * HH;  const float* vb_l = vb + (size_t)l * Hn;
    const float* aow_l = aow + l * HH; const float* aob_l = aob + (size_t)l * Hn;
    const float* iw_l = iw + l * HH;  const float* ib_l = ib + (size_t)l * Hn;
    const float* ow_l = ow + l * HH;  const float* ob_l = ob + (size_t)l * Hn;
    const float* g1_l = g1 + (size_t)l * Hn; const float* b1_l = b1 + (size_t)l * Hn;
    const float* g2_l = g2 + (size_t)l * Hn; const float* b2_l = b2 + (size_t)l * Hn;

    // ---- BERT layer on X (P0) ----
    launch_gemm<EPI_BIAS, false, false>(stream, P0, sAct, HP, qw_l, 0, Hn,
        P2, sAct, HP, HP, Hn, Hn, KS_H, qb_l, nullptr, 0, nullptr, 0.f);          // q
    launch_gemm<EPI_BIAS, false, false>(stream, P0, sAct, HP, kw_l, 0, Hn,
        P3, sAct, HP, HP, Hn, Hn, KS_H, kb_l, nullptr, 0, nullptr, 0.f);          // k
    launch_gemm<EPI_BIAS, false, false>(stream, P0, sAct, HP, vw_l, 0, Hn,
        P4, sAct, HP, HP, Hn, Hn, KS_H, vb_l, nullptr, 0, nullptr, 0.f);          // v
    launch_gemm<EPI_SCORES, true, true>(stream, P2, sAct, HP, P3, sAct, HP,
        S, sNN, Nn, Nn, Nn, HP, KS_H, nullptr, nullptr, 0, maskAdd, scl);         // q@k^T + mask
    softmax_row_kernel<<<Bn * Nn, 256, 0, stream>>>(S);                           // probs
    launch_gemm<EPI_PLAIN, false, true>(stream, S, sNN, Nn, P4, sAct, HP,
        P5, sAct, HP, HP, HP, Nn, KS_N, nullptr, nullptr, 0, nullptr, 0.f);       // ctx
    launch_gemm<EPI_BIAS_RES, false, false>(stream, P5, sAct, HP, aow_l, 0, Hn,
        P6, sAct, HP, HP, Hn, Hn, KS_H, aob_l, P0, sAct, nullptr, 0.f);           // ctx@aow+aob+x
    layernorm_kernel<<<Bn * Nn, 256, 0, stream>>>(P6, P2, g1_l, b1_l);            // attn
    launch_gemm<EPI_BIAS_GELU, false, false>(stream, P2, sAct, HP, iw_l, 0, Hn,
        P6, sAct, HP, HP, Hn, Hn, KS_H, ib_l, nullptr, 0, nullptr, 0.f);          // inter
    launch_gemm<EPI_BIAS_RES, false, false>(stream, P6, sAct, HP, ow_l, 0, Hn,
        P5, sAct, HP, HP, Hn, Hn, KS_H, ob_l, P2, sAct, nullptr, 0.f);            // +attn
    layernorm_kernel<<<Bn * Nn, 256, 0, stream>>>(P5, P3, g2_l, b2_l);            // Xb -> P3

    // ---- GCN on Y (P1) ----
    launch_gemm<EPI_PLAIN, false, false>(stream, P1, sAct, HP, gcn_w, 0, Hn,
        P4, sAct, HP, HP, Hn, Hn, KS_H, nullptr, nullptr, 0, nullptr, 0.f);       // teout
    launch_gemm<EPI_GCN, false, true>(stream, out_adj, sNN, Nn, P4, sAct, HP,
        P5, sAct, HP, HP, HP, Nn, KS_N, nullptr, P1, sAct, denom, 0.f);           // Yg -> P5

    // ---- self-alignment ----
    launch_gemm<EPI_PLAIN, false, false>(stream, P3, sAct, HP, mut_w, 0, Hn,
        P6, sAct, HP, HP, Hn, Hn, KS_H, nullptr, nullptr, 0, nullptr, 0.f);       // t = Xb@mut_w
    launch_gemm<EPI_PAIR, true, true>(stream, P6, sAct, HP, P5, sAct, HP,
        S, sNN, Nn, Nn, Nn, HP, KS_H, nullptr, nullptr, 0, tmask, 0.f);           // logit
    transpose_kernel<<<dim3(Nn / 32, Nn / 32, Bn), dim3(32, 8), 0, stream>>>(S, ST);
    softmax_row_kernel<<<Bn * Nn, 256, 0, stream>>>(ST);                          // p_col^T
    softmax_row_kernel<<<Bn * Nn, 256, 0, stream>>>(S);                           // p_row
    launch_gemm<EPI_ALIGN, false, true>(stream, S, sNN, Nn, P5, sAct, HP,
        P0, sAct, HP, HP, HP, Nn, KS_N, nullptr, P3, sAct, tmask, 0.f);           // new outs -> X
    launch_gemm<EPI_ALIGN, false, true>(stream, ST, sNN, Nn, P3, sAct, HP,
        P1, sAct, HP, HP, HP, Nn, KS_N, nullptr, P5, sAct, tmask, 0.f);           // new output -> Y
  }

  pack_out_kernel<<<(unsigned)(((size_t)Bn * Nn * Hn + 255) / 256), 256, 0, stream>>>(P0, out_outs);
}
